// MyGRULanguageModel_55173149884495
// MI455X (gfx1250) — matrix operand order repaired and checked by reading, not compiled, not run on hardware
//
#include <hip/hip_runtime.h>
#include <hip/hip_bf16.h>
#include <cstdint>
#include <cstddef>

// Problem constants
#define DM    512          // D_MODEL
#define HID   768          // HIDDEN
#define BB    64           // batch
#define SS    512          // seq len
#define NC    2            // classes
#define G3    (3*HID)      // 2304 (gate rows)
#define M_ROWS (SS*BB)     // 32768 (GEMM M)
#define LWP   (HID + 8)    // padded LDS row (bf16 elems) for Wh slice

// ---- WMMA types (CDNA5 / gfx1250, wave32) ----
typedef __attribute__((ext_vector_type(16))) __bf16 v16bf;
typedef __attribute__((ext_vector_type(8)))  float  v8f;

union AB16 { uint4 u[2]; v16bf v; };   // 32B: one 16x32 (A) or 32x16 (B) bf16 fragment per lane

__device__ __forceinline__ unsigned short f2bf(float f) {
    unsigned int u = __float_as_uint(f);
    unsigned int r = (u + 0x7FFFu + ((u >> 16) & 1u)) >> 16;   // round-to-nearest-even
    return (unsigned short)r;
}
__device__ __forceinline__ float bf2f(unsigned short h) {
    return __uint_as_float(((unsigned int)h) << 16);
}
__device__ __forceinline__ float sigmoid_(float x) { return 1.0f / (1.0f + __expf(-x)); }
__device__ __forceinline__ float tanh_(float x)    { return 1.0f - 2.0f / (__expf(2.0f * x) + 1.0f); }

// ---------------------------------------------------------------------------
// 1) f32 -> bf16 conversion (float4 -> ushort4)
// ---------------------------------------------------------------------------
__global__ void k_f32_to_bf16(const float* __restrict__ in, unsigned short* __restrict__ out, int n4) {
    int i = blockIdx.x * blockDim.x + threadIdx.x;
    if (i < n4) {
        float4 f = ((const float4*)in)[i];
        ushort4 o;
        o.x = f2bf(f.x); o.y = f2bf(f.y); o.z = f2bf(f.z); o.w = f2bf(f.w);
        ((ushort4*)out)[i] = o;
    }
}

// ---------------------------------------------------------------------------
// 2) Embedding gather -> x_bf[s*B + b][0..D) bf16 (row-major, K contiguous)
// ---------------------------------------------------------------------------
__global__ void k_gather_embed(const int* __restrict__ ids, const float* __restrict__ emb,
                               unsigned short* __restrict__ x_bf) {
    int row = blockIdx.x;                 // row = s*B + b
    int s = row >> 6;
    int b = row & 63;
    int id = ids[b * SS + s];             // input_ids is [B,S]
    float4 f = ((const float4*)(emb + (size_t)id * DM))[threadIdx.x];
    ushort4 o;
    o.x = f2bf(f.x); o.y = f2bf(f.y); o.z = f2bf(f.z); o.w = f2bf(f.w);
    ((ushort4*)(x_bf + (size_t)row * DM))[threadIdx.x] = o;
}

// ---------------------------------------------------------------------------
// 3) init: zero h buffer 0 (f32 master + bf16 copy) and the grid-barrier ctr
// ---------------------------------------------------------------------------
__global__ void k_init(float* __restrict__ hf, unsigned short* __restrict__ hb,
                       unsigned* __restrict__ ctr, int n) {
    int i = blockIdx.x * blockDim.x + threadIdx.x;
    if (i < n) { hf[i] = 0.0f; hb[i] = 0; }
    if (i == 0) { *ctr = 0u; }
}

// ---------------------------------------------------------------------------
// 4) Input-gate GEMM: xg[dir][m][n] = sum_k x[m][k] * Wi[dir][n][k] + bi[dir][n]
//    M=32768, N=2304, K=512. Each wave: TWO 16x16 N-tiles per A fragment
//    (2 independent WMMA chains, half the LDS A traffic per FLOP).
//    8 waves/block share one M-tile; A strip (16x512 bf16) staged in LDS.
//    B operand = Wi rows read directly (row-major [N,K] == WMMA B layout).
// ---------------------------------------------------------------------------
__global__ void __launch_bounds__(256)
k_xgate_gemm(const unsigned short* __restrict__ x_bf,    // [M_ROWS][DM]
             const unsigned short* __restrict__ wi_bf,   // [2][G3][DM]
             const float* __restrict__ bi_f, const float* __restrict__ bi_b,
             unsigned short* __restrict__ xg_bf)         // [2][M_ROWS][G3]
{
    __shared__ __align__(16) unsigned short lds[16 * (DM + 8)];  // +16B pad/row (bank spread)

    const int mtile = blockIdx.x;                 // 0..2047
    const int dir   = blockIdx.z;                 // 0..1
    const int mbase = mtile * 16;

    // cooperative A-strip load: 16 rows x 512 bf16 = 1024 uint4
    {
        const uint4* src = (const uint4*)(x_bf + (size_t)mbase * DM);
        for (int idx = threadIdx.x; idx < 1024; idx += 256) {
            int r = idx >> 6;                     // 64 uint4 per row
            int c = idx & 63;
            uint4 v = src[r * 64 + c];
            *(uint4*)&lds[r * (DM + 8) + c * 8] = v;
        }
    }
    __syncthreads();

    const int lane = threadIdx.x & 31;
    const int wave = threadIdx.x >> 5;            // 0..7
    const int nt0  = (blockIdx.y * 8 + wave) * 2; // two adjacent N-tiles, 0..142
    const int n0   = nt0 * 16;
    const int n1   = n0 + 16;
    const int r15  = lane & 15;                   // M row (A) / N col (B)
    const int hif  = lane >> 4;                   // lane half

    const unsigned short* wrow0 = wi_bf + ((size_t)dir * G3 + (size_t)(n0 + r15)) * DM;
    const unsigned short* wrow1 = wi_bf + ((size_t)dir * G3 + (size_t)(n1 + r15)) * DM;

    v8f acc0 = {}, acc1 = {};
    for (int k0 = 0; k0 < DM; k0 += 32) {
        AB16 a, b0, b1;
        int ka = k0 + hif * 16;                   // A: contiguous [ka,ka+16)
        a.u[0] = *(const uint4*)&lds[r15 * (DM + 8) + ka];
        a.u[1] = *(const uint4*)&lds[r15 * (DM + 8) + ka + 8];
        int kb = k0 + hif * 16;                   // B: contiguous [kb,kb+16)
        b0.u[0] = *(const uint4*)&wrow0[kb];
        b0.u[1] = *(const uint4*)&wrow0[kb + 8];
        b1.u[0] = *(const uint4*)&wrow1[kb];
        b1.u[1] = *(const uint4*)&wrow1[kb + 8];
        acc0 = __builtin_amdgcn_wmma_f32_16x16x32_bf16(false, a.v, false, b0.v, (short)0, acc0, false, false);
        acc1 = __builtin_amdgcn_wmma_f32_16x16x32_bf16(false, a.v, false, b1.v, (short)0, acc1, false, false);
    }

    const float* bi = dir ? bi_b : bi_f;
    float bias0 = bi[n0 + r15];
    float bias1 = bi[n1 + r15];
    unsigned short* dst = xg_bf + (size_t)dir * M_ROWS * G3;
    #pragma unroll
    for (int i = 0; i < 8; ++i) {                 // C/D layout: M = i + 8*half, N = lane%16
        int m = mbase + i + hif * 8;
        dst[(size_t)m * G3 + (n0 + r15)] = f2bf(acc0[i] + bias0);
        dst[(size_t)m * G3 + (n1 + r15)] = f2bf(acc1[i] + bias1);
    }
}

// ---------------------------------------------------------------------------
// 5) Persistent GRU scan: ONE launch for all 512 timesteps, both directions.
//    grid = (48 j-tiles, 2 dirs) = 96 blocks (co-resident), block = 128
//    (4 waves = 4 M-tiles of B=64). Per block:
//      - Wh slice (3 gates x 16 rows x 768 K bf16 = 72KB) loaded ONCE into
//        LDS via global_load_async_to_lds_b128 (ASYNCcnt), reused 512 steps.
//      - per step: r/z/n tiles = 3 x 24 bf16 WMMAs over K=768 (A = h bf16
//        from L2, B = ds_load_b128), f32 gate math, h ping-pong (f32 master
//        + bf16 copy), prefetch of next step's xg tile, device-wide barrier.
// ---------------------------------------------------------------------------
__global__ void __launch_bounds__(128)
k_gru_scan(const unsigned short* __restrict__ xg_bf,   // [2][S*B][G3]
           const unsigned short* __restrict__ wh_bf,   // [2][G3][HID]
           const float* __restrict__ bh_f, const float* __restrict__ bh_b,
           float* __restrict__ hf,                     // [2 buf][2 dir][B][HID]
           unsigned short* __restrict__ hbf,           // [2 buf][2 dir][B][HID]
           unsigned* __restrict__ bar)
{
    constexpr unsigned NBLK = 96;
    __shared__ __align__(16) unsigned short ldsW[48 * LWP];   // 74,496 B

    const int jt   = blockIdx.x;                  // 0..47
    const int dir  = blockIdx.y;                  // 0..1
    const int j0   = jt * 16;
    const int wave = threadIdx.x >> 5;            // M-tile 0..3
    const int lane = threadIdx.x & 31;
    const int r15  = lane & 15;
    const int hif  = lane >> 4;
    const int mbase = wave * 16;

    // ---- one-time Wh slice -> LDS via async DMA (48 rows x 96 uint4) ----
    {
        const unsigned short* wh = wh_bf + (size_t)dir * G3 * HID;
        const unsigned ldsbase = (unsigned)(uintptr_t)(&ldsW[0]);
        for (int idx = threadIdx.x; idx < 48 * 96; idx += 128) {
            int r = idx / 96;                     // lds row: g*16 + n
            int c = idx - r * 96;                 // uint4 column
            int g = r >> 4, n = r & 15;
            const unsigned short* src = wh + (size_t)(g * HID + j0 + n) * HID + c * 8;
            unsigned dst = ldsbase + (unsigned)((r * LWP + c * 8) * 2);
            asm volatile("global_load_async_to_lds_b128 %0, %1, off"
                         :: "v"(dst), "v"(src) : "memory");
        }
        asm volatile("s_wait_asynccnt 0" ::: "memory");
    }
    __syncthreads();

    const float* bh = dir ? bh_b : bh_f;
    const int j = j0 + r15;
    const float bhr = bh[j], bhz = bh[HID + j], bhn = bh[2 * HID + j];

    const size_t HB  = (size_t)2 * BB * HID;      // elems per h buffer
    const size_t DH  = (size_t)dir * BB * HID;

    for (int t = 0; t < SS; ++t) {
        const unsigned short* hbin = hbf + (size_t)(t & 1) * HB + DH;
        const float*          hfi  = hf  + (size_t)(t & 1) * HB + DH;
        float*                hfo  = hf  + (size_t)((t + 1) & 1) * HB + DH;
        unsigned short*       hbo  = hbf + (size_t)((t + 1) & 1) * HB + DH;
        const unsigned short* hA   = hbin + (size_t)(mbase + r15) * HID;

        v8f ar = {}, az = {}, an = {};
        for (int k0 = 0; k0 < HID; k0 += 32) {
            AB16 a, br_, bz_, bn_;
            int ka = k0 + hif * 16;
            a.u[0] = *(const uint4*)&hA[ka];
            a.u[1] = *(const uint4*)&hA[ka + 8];
            int kb = k0 + hif * 16;
            br_.u[0] = *(const uint4*)&ldsW[(0 * 16 + r15) * LWP + kb];
            br_.u[1] = *(const uint4*)&ldsW[(0 * 16 + r15) * LWP + kb + 8];
            bz_.u[0] = *(const uint4*)&ldsW[(1 * 16 + r15) * LWP + kb];
            bz_.u[1] = *(const uint4*)&ldsW[(1 * 16 + r15) * LWP + kb + 8];
            bn_.u[0] = *(const uint4*)&ldsW[(2 * 16 + r15) * LWP + kb];
            bn_.u[1] = *(const uint4*)&ldsW[(2 * 16 + r15) * LWP + kb + 8];
            ar = __builtin_amdgcn_wmma_f32_16x16x32_bf16(false, a.v, false, br_.v, (short)0, ar, false, false);
            az = __builtin_amdgcn_wmma_f32_16x16x32_bf16(false, a.v, false, bz_.v, (short)0, az, false, false);
            an = __builtin_amdgcn_wmma_f32_16x16x32_bf16(false, a.v, false, bn_.v, (short)0, an, false, false);
        }

        const int trow = dir ? (SS - 1 - t) : t;
        const unsigned short* xg = xg_bf + ((size_t)dir * M_ROWS + (size_t)trow * BB) * G3;

        #pragma unroll
        for (int i = 0; i < 8; ++i) {
            int b = mbase + i + hif * 8;
            size_t xrow = (size_t)b * G3;
            float xr = bf2f(xg[xrow + j]);
            float xz = bf2f(xg[xrow + HID + j]);
            float xn = bf2f(xg[xrow + 2 * HID + j]);
            float hp = hfi[(size_t)b * HID + j];
            float r  = sigmoid_(xr + ar[i] + bhr);
            float z  = sigmoid_(xz + az[i] + bhz);
            float n  = tanh_(xn + r * (an[i] + bhn));
            float hn2 = (1.0f - z) * n + z * hp;
            hfo[(size_t)b * HID + j] = hn2;
            hbo[(size_t)b * HID + j] = f2bf(hn2);
        }

        // warm next step's xg tile while others finish (global_prefetch_b8)
        if (t + 1 < SS) {
            const int tr2 = dir ? (SS - 2 - t) : (t + 1);
            const unsigned short* xg2 = xg_bf + ((size_t)dir * M_ROWS + (size_t)tr2 * BB) * G3;
            for (int idx = threadIdx.x; idx < 192; idx += 128) {    // 64 b-rows x 3 gates
                int g = idx / 64, b = idx - g * 64;
                __builtin_prefetch(xg2 + (size_t)b * G3 + g * HID + j0, 0, 3);
            }
        }

        // ---- device-wide barrier (monotonic counter, 1 per step) ----
        __threadfence();
        __syncthreads();
        if (threadIdx.x == 0) {
            __hip_atomic_fetch_add(bar, 1u, __ATOMIC_RELEASE, __HIP_MEMORY_SCOPE_AGENT);
            const unsigned target = (unsigned)(t + 1) * NBLK;
            while (__hip_atomic_load(bar, __ATOMIC_ACQUIRE, __HIP_MEMORY_SCOPE_AGENT) < target)
                __builtin_amdgcn_s_sleep(1);
        }
        __syncthreads();
    }
}

// ---------------------------------------------------------------------------
// 6) MLP head (tiny): hid = relu([h_fwd,h_bwd] @ W1^T + b1) ; logits = hid@W2^T+b2
// ---------------------------------------------------------------------------
__global__ void k_mlp_hidden(const float* __restrict__ hf0,   // [2][B][HID], buf 0 (final h)
                             const float* __restrict__ W1,    // [HID][2*HID]
                             const float* __restrict__ b1,
                             float* __restrict__ hid)         // [B][HID]
{
    int b = blockIdx.x;
    const float* hF = hf0 + (size_t)b * HID;
    const float* hB = hf0 + (size_t)BB * HID + (size_t)b * HID;
    for (int j = threadIdx.x; j < HID; j += blockDim.x) {
        const float* w = W1 + (size_t)j * (2 * HID);
        float acc = b1[j];
        for (int k = 0; k < HID; ++k) acc += hF[k] * w[k];
        for (int k = 0; k < HID; ++k) acc += hB[k] * w[HID + k];
        hid[(size_t)b * HID + j] = fmaxf(acc, 0.0f);
    }
}

__global__ void k_logits(const float* __restrict__ hid, const float* __restrict__ W2,
                         const float* __restrict__ b2, float* __restrict__ out)
{
    int idx = threadIdx.x;
    if (idx < BB * NC) {
        int b = idx >> 1, c = idx & 1;
        const float* h = hid + (size_t)b * HID;
        const float* w = W2 + (size_t)c * HID;
        float acc = b2[c];
        for (int k = 0; k < HID; ++k) acc += h[k] * w[k];
        out[idx] = acc;
    }
}

// ---------------------------------------------------------------------------
// Launch. Workspace ~349 MB (dominated by bf16 xg: 2*32768*2304*2B).
// ---------------------------------------------------------------------------
extern "C" void kernel_launch(void* const* d_in, const int* in_sizes, int n_in,
                              void* d_out, int out_size, void* d_ws, size_t ws_size,
                              hipStream_t stream)
{
    const int*   ids  = (const int*)  d_in[0];
    const float* emb  = (const float*)d_in[1];
    const float* Wi_f = (const float*)d_in[2];
    const float* Wh_f = (const float*)d_in[3];
    const float* bi_f = (const float*)d_in[4];
    const float* bh_f = (const float*)d_in[5];
    const float* Wi_b = (const float*)d_in[6];
    const float* Wh_b = (const float*)d_in[7];
    const float* bi_b = (const float*)d_in[8];
    const float* bh_b = (const float*)d_in[9];
    const float* W1   = (const float*)d_in[10];
    const float* b1   = (const float*)d_in[11];
    const float* W2   = (const float*)d_in[12];
    const float* b2   = (const float*)d_in[13];
    float* out = (float*)d_out;

    char* p = (char*)d_ws;
    auto take = [&](size_t bytes) -> char* {
        char* q = p;
        p += (bytes + 255) & ~(size_t)255;
        return q;
    };
    unsigned short* x_bf  = (unsigned short*)take((size_t)M_ROWS * DM * 2);       // 33.6 MB
    unsigned short* wi_bf = (unsigned short*)take((size_t)2 * G3 * DM * 2);       // 4.7 MB
    unsigned short* wh_bf = (unsigned short*)take((size_t)2 * G3 * HID * 2);      // 7.1 MB
    unsigned short* xg_bf = (unsigned short*)take((size_t)2 * M_ROWS * G3 * 2);   // 302 MB
    float*          hf    = (float*)         take((size_t)2 * 2 * BB * HID * 4);  // [buf][dir][B][H]
    unsigned short* hbf   = (unsigned short*)take((size_t)2 * 2 * BB * HID * 2);
    float*          hid   = (float*)         take((size_t)BB * HID * 4);
    unsigned*       bar   = (unsigned*)      take(256);

    const int nWi = G3 * DM, nWh = G3 * HID;
    k_f32_to_bf16<<<(nWi / 4 + 255) / 256, 256, 0, stream>>>(Wi_f, wi_bf,       nWi / 4);
    k_f32_to_bf16<<<(nWi / 4 + 255) / 256, 256, 0, stream>>>(Wi_b, wi_bf + nWi, nWi / 4);
    k_f32_to_bf16<<<(nWh / 4 + 255) / 256, 256, 0, stream>>>(Wh_f, wh_bf,       nWh / 4);
    k_f32_to_bf16<<<(nWh / 4 + 255) / 256, 256, 0, stream>>>(Wh_b, wh_bf + nWh, nWh / 4);

    k_gather_embed<<<M_ROWS, 128, 0, stream>>>(ids, emb, x_bf);

    const int nH = 2 * BB * HID;
    k_init<<<(nH + 255) / 256, 256, 0, stream>>>(hf, hbf, bar, nH);

    // big parallel GEMM: grid = (M/16, 144/(8 waves * 2 tiles), 2 dirs)
    k_xgate_gemm<<<dim3(M_ROWS / 16, 9, 2), 256, 0, stream>>>(x_bf, wi_bf, bi_f, bi_b, xg_bf);

    // persistent scan: single launch, 512 steps, fwd+bwd fused
    k_gru_scan<<<dim3(48, 2), 128, 0, stream>>>(xg_bf, wh_bf, bh_f, bh_b, hf, hbf, bar);

    // final h lives in buffer 0 (512 steps -> even parity)
    k_mlp_hidden<<<BB, 256, 0, stream>>>(hf, W1, b1, hid);
    k_logits<<<1, 128, 0, stream>>>(hid, W2, b2, out);
}
